// Res3DBlock_54889682043713
// MI455X (gfx1250) — matrix operand order repaired and checked by reading, not compiled, not run on hardware
//
#include <hip/hip_runtime.h>

// ---------------------------------------------------------------------------
// Res3D sparse-conv block for gfx1250 (MI455X).
// conv1 (27-tap gather GEMM, 64->128) -> BN -> conv2 (27-tap, 128->128) -> BN
// + skip (1x1 GEMM 64->128) -> BN ; relu(sum).
// GEMMs: v_wmma_f32_16x16x32_bf16, A gathered via double-buffered
// global_load_async_to_lds_b128 (ASYNCcnt) so tap t+1's gather overlaps
// tap t's WMMAs.
// ---------------------------------------------------------------------------

#define COUTC 128
#define EPSV 1e-5f

typedef __attribute__((ext_vector_type(16))) __bf16 v16bf;
typedef __attribute__((ext_vector_type(8)))  float  v8f;

// ---------------- small prep kernels ----------------

// converts n elements; elements [n, ntot) are zero-filled (pad row for the
// async gather's sentinel index).
__global__ void cvt_bf16_kernel(const float* __restrict__ in,
                                __bf16* __restrict__ out, int n, int ntot) {
    int i = blockIdx.x * blockDim.x + threadIdx.x;
    if (i >= ntot) return;
    out[i] = (i < n) ? (__bf16)in[i] : (__bf16)0.0f;
}

// in: [T][K][Nc] row-major f32  ->  out: [T][Nc][K] bf16 (column-major per tap)
__global__ void pack_w_kernel(const float* __restrict__ in,
                              __bf16* __restrict__ out, int T, int K, int Nc) {
    int i = blockIdx.x * blockDim.x + threadIdx.x;
    int total = T * K * Nc;
    if (i >= total) return;
    int t = i / (K * Nc);
    int rem = i % (K * Nc);
    int k = rem / Nc;
    int n = rem % Nc;
    out[((size_t)t * Nc + n) * K + k] = (__bf16)in[i];
}

// ---------------- WMMA gather-GEMM with async double buffering ----------------
// out[i, :] = sum_t  A[idx[t][i], :] @ Bw[t]
// A has nrows+1 rows; row nrows is all zeros (sentinel target).
// Block: 256 threads = 8 waves. M-tile 64 (4 sub-tiles of 16); wave w owns a
// 16-column tile. A staged in LDS (2 buffers) by async copies.

template <int CIN, bool GATHER>
__global__ void __launch_bounds__(256)
conv_wmma_kernel(const __bf16* __restrict__ A,   // [(nrows+1)][CIN] bf16
                 const __bf16* __restrict__ Bw,  // [ntaps][128][CIN] bf16 col-major
                 const int*    __restrict__ idx, // [ntaps][nrows] or null
                 float*        __restrict__ out, // [nrows][128] f32
                 int nrows, int ntaps) {
    __shared__ __bf16 lds[2][64 * CIN];

    const int m0   = blockIdx.x * 64;
    const int tid  = threadIdx.x;
    const int lane = tid & 31;
    const int wave = tid >> 5;  // n-tile 0..7
    const int row  = lane & 15;
    const int kh   = lane >> 4; // half-wave selector

    constexpr int CPR = CIN / 8;           // 16B chunks per A row
    constexpr int CPT = (64 * CPR) / 256;  // async issues per thread per tap (2 or 4)

    v8f acc[4];
#pragma unroll
    for (int mt = 0; mt < 4; ++mt)
#pragma unroll
        for (int e = 0; e < 8; ++e) acc[mt][e] = 0.0f;

    // LDS byte offset of the staging buffers (generic ptr low 32 bits == LDS addr)
    const unsigned lds_base = (unsigned)(uintptr_t)(&lds[0][0]);

    auto prefetch = [&](int t, int buf) {
#pragma unroll
        for (int j = 0; j < CPT; ++j) {
            int c   = tid + j * 256;
            int r   = c / CPR;
            int off = c % CPR;
            int gr  = m0 + r;
            int src;
            if (GATHER) {
                src = (gr < nrows) ? idx[(size_t)t * nrows + gr] : nrows;
            } else {
                src = (gr < nrows) ? gr : nrows;  // pad row
            }
            uintptr_t gp = (uintptr_t)(A + (size_t)src * CIN + off * 8);
            unsigned  la = lds_base + (unsigned)(buf * 64 * CIN * 2 + c * 16);
            asm volatile("global_load_async_to_lds_b128 %0, %1, off"
                         :: "v"(la), "v"(gp)
                         : "memory");
        }
    };

    prefetch(0, 0);

    for (int t = 0; t < ntaps; ++t) {
        const int cur = t & 1;
        __syncthreads();  // all waves finished WMMAs on lds[cur^1] (tap t-1)
        if (t + 1 < ntaps) {
            prefetch(t + 1, cur ^ 1);
            // leave only the just-issued prefetch outstanding -> tap t landed
            if constexpr (CPT == 2)
                asm volatile("s_wait_asynccnt 0x2" ::: "memory");
            else
                asm volatile("s_wait_asynccnt 0x4" ::: "memory");
        } else {
            asm volatile("s_wait_asynccnt 0x0" ::: "memory");
        }
        __syncthreads();  // tap t's tile visible block-wide

        const __bf16* abase = &lds[cur][0];
#pragma unroll
        for (int ks = 0; ks < CIN / 32; ++ks) {
            // B fragment: 32x16 tile, lane = column, half-lanes take
            // K {0..7,16..23} vs {8..15,24..31}
            union { v16bf v; uint4 u[2]; } bf;
            const __bf16* bp =
                Bw + ((size_t)t * COUTC + wave * 16 + row) * CIN + ks * 32 + kh * 8;
            bf.u[0] = *(const uint4*)bp;
            bf.u[1] = *(const uint4*)(bp + 16);

#pragma unroll
            for (int mt = 0; mt < 4; ++mt) {
                // A fragment: 16x32 tile; lane = row, half-lanes take
                // K {0..7,16..23} vs {8..15,24..31}
                union { v16bf v; uint4 u[2]; } af;
                const __bf16* ap = abase + (mt * 16 + row) * CIN + ks * 32 + kh * 8;
                af.u[0] = *(const uint4*)ap;
                af.u[1] = *(const uint4*)(ap + 16);

                acc[mt] = __builtin_amdgcn_wmma_f32_16x16x32_bf16(
                    false, af.v, false, bf.v, (short)0, acc[mt], false, false);
            }
        }
    }

    // D layout: vgpr r -> row (kh*8 + r), col = lane&15 within the wave's tile
#pragma unroll
    for (int mt = 0; mt < 4; ++mt) {
        int base = m0 + mt * 16 + kh * 8;
#pragma unroll
        for (int r = 0; r < 8; ++r) {
            int grow = base + r;
            if (grow < nrows)
                out[(size_t)grow * COUTC + wave * 16 + row] = acc[mt][r];
        }
    }
}

// ---------------- deterministic BN statistics ----------------

__global__ void colstats_part_kernel(const float* __restrict__ x, int nrows,
                                     float* __restrict__ part) {
    int col  = threadIdx.x & 127;
    int half = threadIdx.x >> 7;  // 0 or 1
    int rows_per_blk = (nrows + gridDim.x - 1) / gridDim.x;
    int r0 = blockIdx.x * rows_per_blk;
    int r1 = r0 + rows_per_blk;
    if (r1 > nrows) r1 = nrows;
    float s = 0.0f, s2 = 0.0f;
    for (int r = r0 + half; r < r1; r += 2) {
        float v = x[(size_t)r * 128 + col];
        s += v;
        s2 += v * v;
    }
    __shared__ float red[512];
    red[half * 128 + col]       = s;
    red[256 + half * 128 + col] = s2;
    __syncthreads();
    if (half == 0) {
        part[(size_t)blockIdx.x * 256 + col]       = red[col] + red[128 + col];
        part[(size_t)blockIdx.x * 256 + 128 + col] = red[256 + col] + red[384 + col];
    }
}

__global__ void colstats_reduce_kernel(const float* __restrict__ part, int nblk,
                                       float* __restrict__ st) {
    int i = threadIdx.x;  // 0..255
    float s = 0.0f;
    for (int b = 0; b < nblk; ++b) s += part[(size_t)b * 256 + i];
    st[i] = s;
}

// ---------------- BN apply (f32 -> bf16) for conv2 input ----------------
// Also zero-fills the pad row (row nrows).

__global__ void bn_bf16_kernel(const float* __restrict__ x,
                               const float* __restrict__ st,
                               const float* __restrict__ g,
                               const float* __restrict__ b,
                               __bf16* __restrict__ out, int nrows) {
    int i = blockIdx.x * blockDim.x + threadIdx.x;
    if (i >= (nrows + 1) * 128) return;
    if (i >= nrows * 128) { out[i] = (__bf16)0.0f; return; }
    int col    = i & 127;
    float invn = 1.0f / (float)nrows;
    float m    = st[col] * invn;
    float var  = st[128 + col] * invn - m * m;
    float sc   = g[col] * rsqrtf(var + EPSV);
    out[i] = (__bf16)((x[i] - m) * sc + b[col]);
}

// ---------------- fused BN(res) + BN(skip) + add + relu ----------------

__global__ void final_kernel(const float* __restrict__ t2,
                             const float* __restrict__ st2,
                             const float* __restrict__ g2,
                             const float* __restrict__ b2,
                             const float* __restrict__ stsk,
                             const float* __restrict__ gsk,
                             const float* __restrict__ bsk,
                             float* __restrict__ out, int nrows) {
    int i = blockIdx.x * blockDim.x + threadIdx.x;
    if (i >= nrows * 128) return;
    int col    = i & 127;
    float invn = 1.0f / (float)nrows;

    float m2  = st2[col] * invn;
    float v2  = st2[128 + col] * invn - m2 * m2;
    float s2  = g2[col] * rsqrtf(v2 + EPSV);

    float msk = stsk[col] * invn;
    float vsk = stsk[128 + col] * invn - msk * msk;
    float ssk = gsk[col] * rsqrtf(vsk + EPSV);

    float skipv = out[i];
    float r = (t2[i] - m2) * s2 + b2[col] + (skipv - msk) * ssk + bsk[col];
    out[i] = r > 0.0f ? r : 0.0f;
}

// ---------------------------------------------------------------------------

extern "C" void kernel_launch(void* const* d_in, const int* in_sizes, int n_in,
                              void* d_out, int out_size, void* d_ws, size_t ws_size,
                              hipStream_t stream) {
    const float* feats = (const float*)d_in[0];
    const int*   nbr   = (const int*)d_in[1];
    const float* W1    = (const float*)d_in[2];
    const float* g1    = (const float*)d_in[3];
    const float* b1    = (const float*)d_in[4];
    const float* W2    = (const float*)d_in[5];
    const float* g2    = (const float*)d_in[6];
    const float* b2    = (const float*)d_in[7];
    const float* Wsk   = (const float*)d_in[8];
    const float* gsk   = (const float*)d_in[9];
    const float* bsk   = (const float*)d_in[10];
    float* out = (float*)d_out;

    const int nrows = in_sizes[0] / 64;  // 200000

    // workspace carve-out (256B aligned)
    char*  ws  = (char*)d_ws;
    size_t off = 0;
    auto alloc = [&](size_t bytes) -> void* {
        void* p = ws + off;
        off = (off + bytes + 255) & ~(size_t)255;
        return p;
    };
    const int NSTAT_BLK = 256;
    float*  part   = (float*)alloc((size_t)NSTAT_BLK * 256 * 4);
    float*  stats  = (float*)alloc(3 * 256 * 4);               // s1 | s2 | ssk
    __bf16* featsb = (__bf16*)alloc((size_t)(nrows + 1) * 64 * 2);   // + zero pad row
    __bf16* w1t    = (__bf16*)alloc((size_t)27 * 128 * 64 * 2);
    __bf16* w2t    = (__bf16*)alloc((size_t)27 * 128 * 128 * 2);
    __bf16* wskt   = (__bf16*)alloc((size_t)128 * 64 * 2);
    float*  t      = (float*)alloc((size_t)nrows * 128 * 4);   // t1, then t2 (aliased)
    __bf16* hb     = (__bf16*)alloc((size_t)(nrows + 1) * 128 * 2); // + zero pad row
    (void)ws_size; (void)n_in; (void)out_size;

    const int elems64  = nrows * 64;
    const int elems128 = nrows * 128;
    const int gemm_grid = (nrows + 63) / 64;

    // --- precision prep (zero-fills sentinel pad rows) ---
    cvt_bf16_kernel<<<(elems64 + 64 + 255) / 256, 256, 0, stream>>>(
        feats, featsb, elems64, elems64 + 64);
    pack_w_kernel<<<(27 * 64 * 128 + 255) / 256, 256, 0, stream>>>(W1, w1t, 27, 64, 128);
    pack_w_kernel<<<(27 * 128 * 128 + 255) / 256, 256, 0, stream>>>(W2, w2t, 27, 128, 128);
    pack_w_kernel<<<(64 * 128 + 255) / 256, 256, 0, stream>>>(Wsk, wskt, 1, 64, 128);

    // --- conv1 -> t ---
    conv_wmma_kernel<64, true><<<gemm_grid, 256, 0, stream>>>(featsb, w1t, nbr, t, nrows, 27);
    colstats_part_kernel<<<NSTAT_BLK, 256, 0, stream>>>(t, nrows, part);
    colstats_reduce_kernel<<<1, 256, 0, stream>>>(part, NSTAT_BLK, stats);
    bn_bf16_kernel<<<(elems128 + 128 + 255) / 256, 256, 0, stream>>>(t, stats, g1, b1, hb, nrows);

    // --- conv2 -> t (reuse buffer) ---
    conv_wmma_kernel<128, true><<<gemm_grid, 256, 0, stream>>>(hb, w2t, nbr, t, nrows, 27);

    // --- skip GEMM -> d_out ---
    conv_wmma_kernel<64, false><<<gemm_grid, 256, 0, stream>>>(featsb, wskt, nullptr, out, nrows, 1);

    // --- stats for both branches ---
    colstats_part_kernel<<<NSTAT_BLK, 256, 0, stream>>>(t, nrows, part);
    colstats_reduce_kernel<<<1, 256, 0, stream>>>(part, NSTAT_BLK, stats + 256);
    colstats_part_kernel<<<NSTAT_BLK, 256, 0, stream>>>(out, nrows, part);
    colstats_reduce_kernel<<<1, 256, 0, stream>>>(part, NSTAT_BLK, stats + 512);

    // --- fused BN + BN + add + relu (in place on d_out) ---
    final_kernel<<<(elems128 + 255) / 256, 256, 0, stream>>>(
        t, stats + 256, g2, b2, stats + 512, gsk, bsk, out, nrows);
}
